// Quantize_39041252720881
// MI455X (gfx1250) — matrix operand order repaired and checked by reading, not compiled, not run on hardware
//
#include <hip/hip_runtime.h>

// ---------------------------------------------------------------------------
// VQ nearest-codebook quantization for MI455X (gfx1250), bf16x3 WMMA path.
//   x:        (32, 256, 32, 32) f32
//   codebook: (1024, 256)       f32
//   d_out:    [out (N,C,H,W) f32][selected (N,C,H,W) f32]  (concat, 2x 8388608)
// Score s(m,k) = |cb_k|^2 - 2 * dot(x_m, cb_k)  (same argmin as squared dist).
// dot computed as (xh+xl).(ch+cl) ~= xh.ch + xh.cl + xl.ch with f32 WMMA acc.
// ---------------------------------------------------------------------------

typedef __attribute__((ext_vector_type(16))) __bf16 v16bf;
typedef __attribute__((ext_vector_type(8)))  __bf16 v8bf;
typedef __attribute__((ext_vector_type(8)))  float  v8f;

#define C_DIM      256
#define K_CODES    1024
#define HW         1024          // H*W
#define M_TOTAL    32768         // N*H*W
#define BLOCK_ROWS 128           // rows of X per workgroup (8 waves x 16)
#define AS         264           // padded bf16 row stride in LDS (avoids bank conflicts)
#define HALF_OUT   8388608ull    // N*C*H*W

__device__ __forceinline__ void splitf(float x, unsigned short& h, unsigned short& l) {
    unsigned u  = __float_as_uint(x);
    unsigned hu = u & 0xFFFF0000u;          // truncate to bf16 (hi part)
    h = (unsigned short)(hu >> 16);
    float fl = x - __uint_as_float(hu);     // exact residual
    l = (unsigned short)(__float_as_uint(fl) >> 16);
}

// ---- prep: codebook -> bf16 hi/lo + exact f32 squared norms --------------
__global__ void vq_prep_codebook(const float* __restrict__ cb,
                                 unsigned short* __restrict__ cb_hi,
                                 unsigned short* __restrict__ cb_lo,
                                 float* __restrict__ cbnorm) {
    __shared__ float red[256];
    const int k = blockIdx.x;
    const int t = threadIdx.x;
    float v = cb[(size_t)k * C_DIM + t];
    unsigned short h, l;
    splitf(v, h, l);
    cb_hi[(size_t)k * C_DIM + t] = h;
    cb_lo[(size_t)k * C_DIM + t] = l;
    red[t] = v * v;
    __syncthreads();
    for (int s = 128; s > 0; s >>= 1) {
        if (t < s) red[t] += red[t + s];
        __syncthreads();
    }
    if (t == 0) cbnorm[k] = red[0];
}

// ---- main: per-block 128 rows vs all 1024 codes --------------------------
__global__ __launch_bounds__(256, 1)
void vq_main(const float* __restrict__ x,
             const float* __restrict__ cb,            // original f32 codebook
             const unsigned short* __restrict__ cb_hi,
             const unsigned short* __restrict__ cb_lo,
             const float* __restrict__ cbnorm,
             float* __restrict__ out) {
    __shared__ unsigned short AhL[BLOCK_ROWS * AS];   // 66 KB
    __shared__ unsigned short AlL[BLOCK_ROWS * AS];   // 66 KB
    __shared__ float cbnL[K_CODES];                   // 4 KB
    __shared__ int   bestIdxL[BLOCK_ROWS];

    const int tid  = threadIdx.x;
    const int wave = tid >> 5;
    const int lane = tid & 31;
    const int hl   = lane >> 4;     // half-wave (0/1)
    const int ln16 = lane & 15;

    const int m0  = blockIdx.x * BLOCK_ROWS;
    const int n   = m0 >> 10;       // image index (BLOCK_ROWS divides HW)
    const int hw0 = m0 & 1023;

    // ---- stage this block's 128 x-rows into LDS as bf16 hi/lo ----
    {
        const float* xbase = x + ((size_t)n * C_DIM) * HW + hw0;
        for (int i = 0; i < 128; ++i) {
            int flat = i * 256 + tid;          // coalesced over tid
            int r = flat & 127;                // row within block
            int c = flat >> 7;                 // channel
            float v = xbase[(size_t)c * HW + r];
            unsigned short h, l;
            splitf(v, h, l);
            AhL[r * AS + c] = h;
            AlL[r * AS + c] = l;
        }
        for (int i = tid; i < K_CODES; i += 256) cbnL[i] = cbnorm[i];
    }
    __syncthreads();

    // ---- load A fragments (16 rows x 256 c) into registers ----
    // A 16x32 bf16 layout: lane ln16 = M; elems 0..15 -> K = c0..c0+15,
    // with c0 = 32*cc + 16*hl.
    const int rA = wave * 16 + ln16;
    v16bf Ah[8], Al[8];
#pragma unroll
    for (int cc = 0; cc < 8; ++cc) {
        const int c0 = cc * 32 + 16 * hl;
        union { v16bf v; v8bf h2[2]; } ua, ub;
        ua.h2[0] = *(const v8bf*)(&AhL[rA * AS + c0]);
        ua.h2[1] = *(const v8bf*)(&AhL[rA * AS + c0 + 8]);
        Ah[cc] = ua.v;
        ub.h2[0] = *(const v8bf*)(&AlL[rA * AS + c0]);
        ub.h2[1] = *(const v8bf*)(&AlL[rA * AS + c0 + 8]);
        Al[cc] = ub.v;
    }

    float bestv[8];
    int   besti[8];
#pragma unroll
    for (int v = 0; v < 8; ++v) { bestv[v] = 3.4e38f; besti[v] = 0; }

    // ---- loop over 64 code tiles of 16 ----
    // B 32x16 bf16 layout: lane ln16 = N(code); elems = 16 consecutive K(c)
    // starting at 32*cc + 16*hl. Row-major split codebook gives each lane a
    // contiguous 32B read (L2-resident, shared by all waves/blocks).
    for (int kt = 0; kt < K_CODES / 16; ++kt) {
        const unsigned short* rowH = cb_hi + ((size_t)(kt * 16 + ln16)) * C_DIM;
        const unsigned short* rowL = cb_lo + ((size_t)(kt * 16 + ln16)) * C_DIM;
        if (kt + 1 < K_CODES / 16) {
            __builtin_prefetch(rowH + 16 * C_DIM, 0, 0);
            __builtin_prefetch(rowL + 16 * C_DIM, 0, 0);
        }
        v8f acc = {};
#pragma unroll
        for (int cc = 0; cc < 8; ++cc) {
            const int cB = cc * 32 + 16 * hl;
            v16bf bh = *(const v16bf*)(rowH + cB);   // 32B aligned
            v16bf bl = *(const v16bf*)(rowL + cB);
            acc = __builtin_amdgcn_wmma_f32_16x16x32_bf16(false, Ah[cc], false, bh,
                                                          (short)0, acc, false, false);
            acc = __builtin_amdgcn_wmma_f32_16x16x32_bf16(false, Ah[cc], false, bl,
                                                          (short)0, acc, false, false);
            acc = __builtin_amdgcn_wmma_f32_16x16x32_bf16(false, Al[cc], false, bh,
                                                          (short)0, acc, false, false);
        }
        // C/D layout: lane -> N = ln16, VGPR v -> M = v + 8*hl
        const int nglob = kt * 16 + ln16;
        const float cbn = cbnL[nglob];
#pragma unroll
        for (int v = 0; v < 8; ++v) {
            float s = fmaf(-2.0f, acc[v], cbn);
            if (s < bestv[v]) { bestv[v] = s; besti[v] = nglob; }  // kt ascending => first-min
        }
    }

    // ---- cross-lane argmin within each 16-lane half ----
#pragma unroll
    for (int v = 0; v < 8; ++v) {
        float bv = bestv[v];
        int   bi = besti[v];
        for (int mask = 8; mask >= 1; mask >>= 1) {
            float ov = __shfl_xor(bv, mask, 32);
            int   oi = __shfl_xor(bi, mask, 32);
            if (ov < bv || (ov == bv && oi < bi)) { bv = ov; bi = oi; }
        }
        if (ln16 == 0) bestIdxL[wave * 16 + v + 8 * hl] = bi;  // row m = v + 8*hl
    }
    __syncthreads();

    // ---- gather selected codebook rows, write both output halves ----
    {
        const int r  = tid & 127;
        const int cp = tid >> 7;
        const int code = bestIdxL[r];
        const float* cbrow = cb + (size_t)code * C_DIM;
        const size_t base = ((size_t)n * C_DIM) * HW + hw0 + r;
        for (int c = cp; c < C_DIM; c += 2) {
            float sel = cbrow[c];
            size_t off = base + (size_t)c * HW;      // coalesced over r
            float xv = x[off];
            out[off]            = xv + (sel - xv);   // faithful to reference expr
            out[HALF_OUT + off] = sel;
        }
    }
}

// ---------------------------------------------------------------------------
extern "C" void kernel_launch(void* const* d_in, const int* in_sizes, int n_in,
                              void* d_out, int out_size, void* d_ws, size_t ws_size,
                              hipStream_t stream) {
    (void)in_sizes; (void)n_in; (void)out_size; (void)ws_size;
    const float* x  = (const float*)d_in[0];
    const float* cb = (const float*)d_in[1];

    unsigned short* cb_hi = (unsigned short*)d_ws;                 // 512 KB
    unsigned short* cb_lo = cb_hi + (size_t)K_CODES * C_DIM;       // 512 KB
    float*          cbnrm = (float*)(cb_lo + (size_t)K_CODES * C_DIM); // 4 KB
    float* out = (float*)d_out;

    vq_prep_codebook<<<K_CODES, 256, 0, stream>>>(cb, cb_hi, cb_lo, cbnrm);
    vq_main<<<M_TOTAL / BLOCK_ROWS, 256, 0, stream>>>(x, cb, cb_hi, cb_lo, cbnrm, out);
}
